// Fushion_75290776699450
// MI455X (gfx1250) — compile-verified, matrix operand order checked by reading, not run on hardware
//
#include <hip/hip_runtime.h>
#include <stdint.h>

typedef __attribute__((ext_vector_type(16))) __bf16 v16bf;
typedef __attribute__((ext_vector_type(8)))  float  v8f;
typedef __attribute__((ext_vector_type(4)))  unsigned int u32x4;
typedef __attribute__((ext_vector_type(8)))  unsigned int u32x8;
typedef __attribute__((ext_vector_type(4)))  int i32x4;
typedef __attribute__((ext_vector_type(8)))  int i32x8;

#define BB   16
#define IH   32
#define IW   64
#define HW   2048
#define CIN  512
#define NCH  512
#define KTOT 4608
#define BK   32
#define NK   (KTOT / BK)
#define LDA  40          /* padded halves per LDS row: 80B, 16B aligned, conflict-free */
#define PP   16
#define CC   8

__device__ __forceinline__ unsigned short f2bf(float f) {
  union { float f; unsigned int u; } v; v.f = f;
  unsigned int u = v.u;
  unsigned int r = u + 0x7FFFu + ((u >> 16) & 1u);     // round-to-nearest-even
  if ((u & 0x7F800000u) == 0x7F800000u) r = u;         // inf/nan passthrough
  return (unsigned short)(r >> 16);
}

// ---------------- prep: encoder NCHW f32 -> NHWC bf16 ----------------
__global__ __launch_bounds__(256) void k_prep_enc(const float* __restrict__ enc,
                                                  unsigned short* __restrict__ encB) {
  int i = blockIdx.x * 256 + threadIdx.x;              // over B*HW*CIN
  if (i >= BB * HW * CIN) return;
  int c   = i & (CIN - 1);
  int pix = (i >> 9) & (HW - 1);
  int b   = i >> 20;
  encB[i] = f2bf(enc[((size_t)(b * CIN + c)) * HW + pix]);
}

// ------------- prep: weights -> [Cout][K=(tap*512+cin)] bf16, BN fold -------------
__global__ __launch_bounds__(256) void k_prep_w(const float* __restrict__ fw,
                                                const float* __restrict__ g,
                                                const float* __restrict__ be,
                                                const float* __restrict__ mn,
                                                const float* __restrict__ vr,
                                                unsigned short* __restrict__ wB,
                                                float* __restrict__ bnS,
                                                float* __restrict__ bnH) {
  int i = blockIdx.x * 256 + threadIdx.x;
  if (i < NCH * KTOT) {
    int n = i / KTOT;
    int k = i - n * KTOT;
    int tap = k >> 9;
    int cin = k & 511;
    wB[i] = f2bf(fw[(size_t)(n * CIN + cin) * 9 + tap]);
  }
  if (i < NCH) {
    float inv = g[i] * rsqrtf(vr[i] + 1e-5f);
    bnS[i] = inv;
    bnH[i] = be[i] - mn[i] * inv;
  }
}

// ---------------- implicit-GEMM conv3x3 512->512 with WMMA bf16 + TDM ----------------
__global__ __launch_bounds__(256) void k_gemm(const unsigned short* __restrict__ encB,
                                              const unsigned short* __restrict__ wB,
                                              const float* __restrict__ bnS,
                                              const float* __restrict__ bnH,
                                              float* __restrict__ feat) {
  __shared__ alignas(16) unsigned short As[2][128][LDA];   // channels x k (TDM-filled)
  __shared__ alignas(16) unsigned short Bs[2][128][LDA];   // pixels   x k

  const int t     = threadIdx.x;
  const int lane  = t & 31;
  const int wave  = t >> 5;
  const int waveM = wave & 3;                // 4 waves along channels
  const int waveN = wave >> 2;               // 2 waves along pixels
  const int chT   = blockIdx.x * 128;
  const int pxT   = blockIdx.y * 128;

  const int frow = lane & 15;
  const int fgrp = lane >> 4;
  const int fc0  = fgrp * 8;                 // K 0..7 / 8..15
  const int fc1  = 16 + fgrp * 8;            // K 16..23 / 24..31

  v8f acc[2][4] = {};
  u32x4 rb[2];

  // --- TDM: weight tile [128 ch][32 bf16], row stride 4608 elems, LDS rows padded
  //     to 80B via pad_enable (interval = 16 DWORDs stored, amount = 4 DWORDs).
  auto tdmLoadA = [&](int kt, int stg) {
    if (wave != 0) return;                   // one issuing wave per workgroup
    unsigned long long ga = (unsigned long long)(uintptr_t)wB
        + (((unsigned long long)chT * KTOT) + (unsigned long long)kt * BK) * 2ull;
    unsigned int lds = (unsigned int)(uintptr_t)&As[stg][0][0];
    u32x4 g0;
    g0[0] = 1u;                                            // count=1, user mode
    g0[1] = lds;                                           // lds_addr (bytes)
    g0[2] = (unsigned int)ga;                              // global_addr[31:0]
    g0[3] = ((unsigned int)(ga >> 32) & 0x01FFFFFFu)       // global_addr[56:32]
          | (2u << 30);                                    // type = 2 ("image")
    i32x8 g1;
    g1[0] = (int)((1u << 16)      // data_size = 2 bytes
                | (1u << 20)      // pad_enable
                | (3u << 22)      // pad_interval: every 16 DWORDs (64B row)
                | (3u << 25));    // pad_amount: 4 DWORDs (16B)
    g1[1] = (int)((KTOT & 0xFFFFu) << 16);                 // tensor_dim0 lo16
    g1[2] = (int)(((KTOT >> 16) & 0xFFFFu)                 // tensor_dim0 hi16
                | ((NCH & 0xFFFFu) << 16));                // tensor_dim1 lo16
    g1[3] = (int)(((NCH >> 16) & 0xFFFFu) | (BK << 16));   // dim1 hi16 | tile_dim0=32
    g1[4] = 128;                                           // tile_dim1=128, tile_dim2=0
    g1[5] = KTOT;                                          // tensor_dim0_stride lo32
    g1[6] = 0;                                             // stride hi16 | dim1_stride
    g1[7] = 0;
    i32x4 z4 = {};
#if __has_include(<hip/amd_detail/amd_gfx1250_TDM.h>)
    i32x8 z8 = {};
    __builtin_amdgcn_tensor_load_to_lds(g0, g1, z4, z4, z8, 0);
#else
    __builtin_amdgcn_tensor_load_to_lds(g0, g1, z4, z4, 0);
#endif
  };

  auto loadB = [&](int kt) {
    int k0   = kt * BK;
    int tap  = k0 >> 9;
    int cin0 = k0 & 511;
    int ky = tap / 3 - 1, kx = tap % 3 - 1;
#pragma unroll
    for (int h = 0; h < 2; ++h) {
      int q   = t + h * 256;
      int row = q >> 2;
      int c8  = (q & 3) * 8;
      int gp = pxT + row;
      int b = gp >> 11, pix = gp & 2047;
      int y = pix >> 6, x = pix & 63;
      int yy = y + ky, xx = x + kx;
      u32x4 z = {};
      if (yy >= 0 && yy < IH && xx >= 0 && xx < IW)
        z = *(const u32x4*)(encB + (size_t)(b * HW + yy * IW + xx) * CIN + cin0 + c8);
      rb[h] = z;
    }
  };
  auto storeB = [&](int st) {
#pragma unroll
    for (int h = 0; h < 2; ++h) {
      int q   = t + h * 256;
      int row = q >> 2;
      int c8  = (q & 3) * 8;
      *(u32x4*)(&Bs[st][row][c8]) = rb[h];
    }
  };

  tdmLoadA(0, 0);
  loadB(0);
  storeB(0);
  __builtin_amdgcn_s_wait_tensorcnt(0);
  __syncthreads();

#pragma unroll 2
  for (int kt = 0; kt < NK; ++kt) {
    int st = kt & 1;
    if (kt + 1 < NK) {
      tdmLoadA(kt + 1, st ^ 1);              // async DMA overlaps compute below
      loadB(kt + 1);
      if (kt + 2 < NK)                       // warm GL2 for the TDM weight stream
        __builtin_prefetch(wB + (size_t)(chT + (t & 127)) * KTOT + (kt + 2) * BK, 0, 1);
    }
    // compute from stage st
    {
      v16bf afr[2], bfr[4];
#pragma unroll
      for (int i = 0; i < 2; ++i) {
        int arow = waveM * 32 + i * 16 + frow;
        u32x4 lo = *(const u32x4*)&As[st][arow][fc0];
        u32x4 hi = *(const u32x4*)&As[st][arow][fc1];
        u32x8 r;
        r[0]=lo[0]; r[1]=lo[1]; r[2]=lo[2]; r[3]=lo[3];
        r[4]=hi[0]; r[5]=hi[1]; r[6]=hi[2]; r[7]=hi[3];
        afr[i] = __builtin_bit_cast(v16bf, r);
      }
#pragma unroll
      for (int j = 0; j < 4; ++j) {
        int brow = waveN * 64 + j * 16 + frow;
        u32x4 lo = *(const u32x4*)&Bs[st][brow][fc0];
        u32x4 hi = *(const u32x4*)&Bs[st][brow][fc1];
        u32x8 r;
        r[0]=lo[0]; r[1]=lo[1]; r[2]=lo[2]; r[3]=lo[3];
        r[4]=hi[0]; r[5]=hi[1]; r[6]=hi[2]; r[7]=hi[3];
        bfr[j] = __builtin_bit_cast(v16bf, r);
      }
#pragma unroll
      for (int i = 0; i < 2; ++i)
#pragma unroll
        for (int j = 0; j < 4; ++j)
          acc[i][j] = __builtin_amdgcn_wmma_f32_16x16x32_bf16(
              false, afr[i], false, bfr[j], (short)0, acc[i][j], false, false);
    }
    if (kt + 1 < NK) storeB(st ^ 1);
    __builtin_amdgcn_s_wait_tensorcnt(0);    // no-op for non-issuing waves
    __syncthreads();
  }

  // epilogue: BN + ReLU, store NCHW (16 consecutive pixels per lane group)
#pragma unroll
  for (int i = 0; i < 2; ++i) {
#pragma unroll
    for (int j = 0; j < 4; ++j) {
      int chBase = chT + waveM * 32 + i * 16 + 8 * fgrp;
      int gp = pxT + waveN * 64 + j * 16 + (lane & 15);
      int b = gp >> 11, pix = gp & 2047;
#pragma unroll
      for (int r = 0; r < 8; ++r) {
        int ch = chBase + r;
        float v = acc[i][j][r] * bnS[ch] + bnH[ch];
        feat[((size_t)(b * NCH + ch)) * HW + pix] = fmaxf(v, 0.0f);
      }
    }
  }
}

// ---------------- neighbor-pair depthwise 5x5 fusion (scatter-add form) ----------------
__constant__ int c_pairbase[16] = {0,2,5,9,13,17,21,25,29,33,37,41,45,49,53,56};

__global__ __launch_bounds__(256) void k_fusion(const float* __restrict__ feat,
                                                const float* __restrict__ fw,
                                                const float* __restrict__ fb,
                                                float* __restrict__ fus) {
  int i = blockIdx.x * 256 + threadIdx.x;              // over B*512*HW
  if (i >= BB * NCH * HW) return;
  int pix = i & 2047;
  int ch  = (i >> 11) & 511;
  int b   = i >> 20;
  int x = pix & 63, y = pix >> 6;
  int c = ch & 7, p = (ch >> 3) & 15, k = ch >> 7;

  float acc = feat[i];
  int pairIdx = c_pairbase[p];
#pragma unroll
  for (int d = 0; d < 4; ++d) {
    int delta = (d < 2) ? (d - 2) : (d - 1);           // -2,-1,1,2 in reference order
    bool valid = (delta == -2) ? (p >= 2) : (delta == -1) ? (p >= 1)
               : (delta == 1)  ? (p <= 14) : (p <= 13);
    if (!valid) continue;
    int src = p + delta;
    const float* plane = feat + ((size_t)b * NCH + (k * 128 + src * 8 + c)) * HW;
    const float* wk = fw + (size_t)(pairIdx * CC + c) * 25;
    float s = fb[pairIdx * CC + c];
#pragma unroll
    for (int ty = 0; ty < 5; ++ty) {
      int yy = y + ty - 2;
      if (yy < 0 || yy >= IH) continue;
#pragma unroll
      for (int tx = 0; tx < 5; ++tx) {
        int xx = x + tx - 2;
        if (xx < 0 || xx >= IW) continue;
        s += plane[yy * IW + xx] * wk[ty * 5 + tx];
      }
    }
    acc += s;
    pairIdx++;
  }
  fus[i] = acc;
}

// ---------------- channel mean/max pooling over 128 channels per branch ----------------
__global__ __launch_bounds__(256) void k_pool(const float* __restrict__ fus,
                                              float* __restrict__ pooled) {
  int i = blockIdx.x * 256 + threadIdx.x;              // over B*4*HW
  if (i >= BB * 4 * HW) return;
  int pix = i & 2047;
  int k = (i >> 11) & 3;
  int b = i >> 13;
  const float* base = fus + ((size_t)b * NCH + k * 128) * HW + pix;
  float s = 0.f, m = -3.0e38f;
  for (int c = 0; c < 128; ++c) { float v = base[(size_t)c * HW]; s += v; m = fmaxf(m, v); }
  pooled[((size_t)b * 8 + k * 2 + 0) * HW + pix] = s * (1.0f / 128.0f);
  pooled[((size_t)b * 8 + k * 2 + 1) * HW + pix] = m;
}

// ---------------- grouped 3x3 conv: 2 -> 1 per branch ----------------
__global__ __launch_bounds__(256) void k_allf(const float* __restrict__ pooled,
                                              const float* __restrict__ gw,
                                              float* __restrict__ allf) {
  int i = blockIdx.x * 256 + threadIdx.x;              // over B*4*HW
  if (i >= BB * 4 * HW) return;
  int pix = i & 2047;
  int k = (i >> 11) & 3;
  int b = i >> 13;
  int x = pix & 63, y = pix >> 6;
  float s = 0.f;
  for (int ic = 0; ic < 2; ++ic) {
    const float* pl = pooled + ((size_t)b * 8 + k * 2 + ic) * HW;
    const float* wk = gw + (k * 2 + ic) * 9;
#pragma unroll
    for (int ty = 0; ty < 3; ++ty) {
      int yy = y + ty - 1;
      if (yy < 0 || yy >= IH) continue;
#pragma unroll
      for (int tx = 0; tx < 3; ++tx) {
        int xx = x + tx - 1;
        if (xx < 0 || xx >= IW) continue;
        s += pl[yy * IW + xx] * wk[ty * 3 + tx];
      }
    }
  }
  allf[((size_t)b * 4 + k) * HW + pix] = s;
}

// ---------------- 4x4 gram attention over global maps ----------------
__global__ __launch_bounds__(256) void k_att(const float* __restrict__ allf,
                                             float* __restrict__ fg) {
  __shared__ float red[256][16];
  __shared__ float wmat[16];
  int b = blockIdx.x, t = threadIdx.x;
  const float* fa = allf + (size_t)b * 4 * HW;
  float ps[16];
  for (int j = 0; j < 16; ++j) ps[j] = 0.f;
  for (int n = t; n < HW; n += 256) {
    float v[4] = { fa[n], fa[HW + n], fa[2 * HW + n], fa[3 * HW + n] };
    for (int c = 0; c < 4; ++c)
      for (int d = 0; d < 4; ++d) ps[c * 4 + d] += v[c] * v[d];
  }
  for (int j = 0; j < 16; ++j) red[t][j] = ps[j];
  __syncthreads();
  for (int s = 128; s > 0; s >>= 1) {
    if (t < s) for (int j = 0; j < 16; ++j) red[t][j] += red[t + s][j];
    __syncthreads();
  }
  if (t < 4) {
    float a[4]; for (int d = 0; d < 4; ++d) a[d] = red[0][t * 4 + d];
    float rmax = fmaxf(fmaxf(a[0], a[1]), fmaxf(a[2], a[3]));
    float l[4], lm = -3.0e38f;
    for (int d = 0; d < 4; ++d) { l[d] = rmax - a[d]; lm = fmaxf(lm, l[d]); }
    float e[4], se = 0.f;
    for (int d = 0; d < 4; ++d) { e[d] = expf(l[d] - lm); se += e[d]; }
    for (int d = 0; d < 4; ++d) wmat[t * 4 + d] = e[d] / se;
  }
  __syncthreads();
  float* fgb = fg + (size_t)b * 4 * HW;
  for (int n = t; n < HW; n += 256) {
    float v[4] = { fa[n], fa[HW + n], fa[2 * HW + n], fa[3 * HW + n] };
    for (int c = 0; c < 4; ++c) {
      float s = v[c];
      for (int d = 0; d < 4; ++d) s += wmat[c * 4 + d] * v[d];
      fgb[c * HW + n] = s;
    }
  }
}

// ---------------- 1x1 heatmap heads + flat softmax + DSNT ----------------
__global__ __launch_bounds__(256) void k_hm(const float* __restrict__ fus,
                                            const float* __restrict__ feat,
                                            const float* __restrict__ fg,
                                            const float* __restrict__ hmw,
                                            const float* __restrict__ hmb,
                                            const float* __restrict__ ow,
                                            const float* __restrict__ ob,
                                            float* __restrict__ out) {
  __shared__ float logit[HW];
  __shared__ float r0[256], r1[256], r2[256];
  int idx = blockIdx.x;
  int p = idx & 15;
  int k = (idx >> 4) & 3;
  int b = (idx >> 6) & 15;
  int which = idx >> 10;                               // 0: fused head, 1: origin head
  int t = threadIdx.x;
  const float* base = (which == 0 ? fus : feat) + ((size_t)b * NCH + k * 128 + p * 8) * HW;
  const float* fgp = fg + ((size_t)b * 4 + k) * HW;
  float wv[CC], bias, g0 = 0.f;
  if (which == 0) {
    const float* h = hmw + (k * PP + p) * (CC + 1);
    g0 = h[0];
    for (int c = 0; c < CC; ++c) wv[c] = h[1 + c];
    bias = hmb[k * PP + p];
  } else {
    const float* h = ow + (k * PP + p) * CC;
    for (int c = 0; c < CC; ++c) wv[c] = h[c];
    bias = ob[k * PP + p];
  }
  float lmax = -3.0e38f;
  for (int n = t; n < HW; n += 256) {
    float s = bias;
    if (which == 0) s += fgp[n] * g0;
    for (int c = 0; c < CC; ++c) s += base[(size_t)c * HW + n] * wv[c];
    logit[n] = s;
    lmax = fmaxf(lmax, s);
  }
  r0[t] = lmax; __syncthreads();
  for (int s = 128; s > 0; s >>= 1) { if (t < s) r0[t] = fmaxf(r0[t], r0[t + s]); __syncthreads(); }
  float M = r0[0];
  __syncthreads();
  float se = 0.f, sx = 0.f, sy = 0.f;
  for (int n = t; n < HW; n += 256) {
    float e = expf(logit[n] - M);
    int x = n & 63, y = n >> 6;
    se += e;
    sx += e * ((2.0f * x + 1.0f) / 64.0f - 1.0f);
    sy += e * ((2.0f * y + 1.0f) / 32.0f - 1.0f);
  }
  r0[t] = se; r1[t] = sx; r2[t] = sy; __syncthreads();
  for (int s = 128; s > 0; s >>= 1) {
    if (t < s) { r0[t] += r0[t + s]; r1[t] += r1[t + s]; r2[t] += r2[t + s]; }
    __syncthreads();
  }
  if (t == 0) {
    int sidx = which * 4 + k;
    size_t o = (((size_t)sidx * BB + b) * PP + p) * 2;
    out[o + 0] = r1[0] / r0[0];
    out[o + 1] = r2[0] / r0[0];
  }
}

extern "C" void kernel_launch(void* const* d_in, const int* in_sizes, int n_in,
                              void* d_out, int out_size, void* d_ws, size_t ws_size,
                              hipStream_t stream) {
  (void)in_sizes; (void)n_in; (void)out_size; (void)ws_size;
  const float* enc = (const float*)d_in[0];
  const float* fw  = (const float*)d_in[1];
  const float* bng = (const float*)d_in[2];
  const float* bnb = (const float*)d_in[3];
  const float* bnm = (const float*)d_in[4];
  const float* bnv = (const float*)d_in[5];
  const float* fuw = (const float*)d_in[6];
  const float* fub = (const float*)d_in[7];
  const float* gw  = (const float*)d_in[8];
  const float* hmw = (const float*)d_in[9];
  const float* hmb = (const float*)d_in[10];
  const float* ow  = (const float*)d_in[11];
  const float* ob  = (const float*)d_in[12];
  float* outp = (float*)d_out;

  char* ws = (char*)d_ws;
  unsigned short* encB = (unsigned short*)(ws + 0);             // 33,554,432 B
  unsigned short* wB   = (unsigned short*)(ws + 33554432);      //  4,718,592 B
  float* bnS   = (float*)(ws + 38273024);                       //      2,048 B
  float* bnH   = (float*)(ws + 38275072);                       //      2,048 B
  float* feat  = (float*)(ws + 38277120);                       // 67,108,864 B
  float* fus   = (float*)(ws + 105385984);                      // 67,108,864 B
  float* pool  = (float*)(ws + 172494848);                      //  1,048,576 B
  float* allf  = (float*)(ws + 173543424);                      //    524,288 B
  float* fg    = (float*)(ws + 174067712);                      //    524,288 B

  k_prep_enc<<<dim3(65536), dim3(256), 0, stream>>>(enc, encB);
  k_prep_w  <<<dim3(9216),  dim3(256), 0, stream>>>(fw, bng, bnb, bnm, bnv, wB, bnS, bnH);
  k_gemm    <<<dim3(4, 256), dim3(256), 0, stream>>>(encB, wB, bnS, bnH, feat);
  k_fusion  <<<dim3(65536), dim3(256), 0, stream>>>(feat, fuw, fub, fus);
  k_pool    <<<dim3(512),   dim3(256), 0, stream>>>(fus, pool);
  k_allf    <<<dim3(512),   dim3(256), 0, stream>>>(pool, gw, allf);
  k_att     <<<dim3(16),    dim3(256), 0, stream>>>(allf, fg);
  k_hm      <<<dim3(2048),  dim3(256), 0, stream>>>(fus, feat, fg, hmw, hmb, ow, ob, outp);
}
